// KGAT_6227702579355
// MI455X (gfx1250) — compile-verified, matrix operand order checked by reading, not run on hardware
//
#include <hip/hip_runtime.h>
#include <hip/hip_bf16.h>

typedef __attribute__((ext_vector_type(16))) _Float16 v16h;
typedef __attribute__((ext_vector_type(8)))  float    v8f;

#define N_USERS 50000
#define N_ENT   100000
#define N_NODES (N_USERS + N_ENT)
#define N_EDGES 2400000
#define EMB     64
#define OUT_COLS 176  // 64 + 64 + 32 + 16

// ---------------------------------------------------------------------------
// Kernel 1: concat(user_embed, entity_embed) -> x0, also write cols [0,64) of out
// ---------------------------------------------------------------------------
__global__ __launch_bounds__(256)
void init_x_kernel(const float* __restrict__ ue, const float* __restrict__ ee,
                   float* __restrict__ x, float* __restrict__ out)
{
    int gid = blockIdx.x * blockDim.x + threadIdx.x;   // over N_NODES * 16 (float4)
    if (gid >= N_NODES * (EMB / 4)) return;
    int row = gid >> 4;
    int c4  = (gid & 15) << 2;
    const float* src = (row < N_USERS) ? (ue + (size_t)row * EMB)
                                       : (ee + (size_t)(row - N_USERS) * EMB);
    float4 v = *(const float4*)(src + c4);
    *(float4*)(x + (size_t)row * EMB + c4) = v;
    *(float4*)(out + (size_t)row * OUT_COLS + c4) = v;
}

// ---------------------------------------------------------------------------
// Kernel 2: pack a (din x dout) row-major f32 weight matrix into per-lane
// f16 B-fragments matching the v_wmma_f32_16x16x32_f16 operand layout.
// grid = (din/32)*(dout/16) fragments, block = 32 lanes.
// Fragment storage: dst[(frag*32 + lane)*16 + e]
// ---------------------------------------------------------------------------
__global__ __launch_bounds__(32)
void pack_weights_kernel(const float* __restrict__ W, _Float16* __restrict__ dst,
                         int din, int dout)
{
    int nt   = dout >> 4;
    int frag = blockIdx.x;
    int kt   = frag / nt;
    int t    = frag % nt;
    int lane = threadIdx.x & 31;
    int nl   = lane & 15;
    int kh   = lane >> 4;
    int n    = t * 16 + nl;

    _Float16* p = dst + ((size_t)frag * 32 + lane) * 16;
#pragma unroll
    for (int e = 0; e < 8; ++e) {
        int k0 = kt * 32 + kh * 8 + e;        // halves 0..7
        int k1 = k0 + 16;                     // halves 8..15
        p[e]     = (_Float16)W[(size_t)k0 * dout + n];
        p[e + 8] = (_Float16)W[(size_t)k1 * dout + n];
    }
}

// ---------------------------------------------------------------------------
// Kernel 3: SpMM scatter:  side[row,:] += edge_val * x[col,:]
// One thread handles 4 contiguous channels of one edge (float4 gather,
// 4x global_atomic_add_f32 scatter; both L2-resident on MI455X).
// ---------------------------------------------------------------------------
template <int DIN>
__global__ __launch_bounds__(256)
void spmm_scatter_kernel(const float* __restrict__ x, const float* __restrict__ ev,
                         const int* __restrict__ erow, const int* __restrict__ ecol,
                         float* __restrict__ side)
{
    constexpr int CH = DIN / 4;
    int gid = blockIdx.x * blockDim.x + threadIdx.x;
    if (gid >= N_EDGES * CH) return;
    int e = gid / CH;
    int c = (gid % CH) * 4;
    float v  = ev[e];
    int col  = ecol[e];
    int row  = erow[e];
    float4 xs = *(const float4*)(x + (size_t)col * DIN + c);
    float* dst = side + (size_t)row * DIN + c;
    atomicAdd(dst + 0, v * xs.x);
    atomicAdd(dst + 1, v * xs.y);
    atomicAdd(dst + 2, v * xs.z);
    atomicAdd(dst + 3, v * xs.w);
}

// ---------------------------------------------------------------------------
// Kernel 4: fused layer:
//   out_row = L2norm( lrelu((x+side)@Wgc + bgc) + lrelu((x*side)@Wbi + bbi) )
// One wave32 per 16-node tile; WMMA f32_16x16x32_f16 over KT k-steps, NT n-tiles.
// Weights are staged into LDS once per 256-thread block (all 8 waves share
// them), so WMMA B operands come from low-latency ds_load_b128 instead of
// per-wave VMEM round-trips.
// ---------------------------------------------------------------------------
template <int DIN, int DOUT>
__global__ __launch_bounds__(256)
void layer_kernel(const float* __restrict__ x, const float* __restrict__ side,
                  const _Float16* __restrict__ Wgc, const float* __restrict__ bgc,
                  const _Float16* __restrict__ Wbi, const float* __restrict__ bbi,
                  float* __restrict__ x_next, float* __restrict__ out)
{
    constexpr int KT = DIN / 32;
    constexpr int NT = DOUT / 16;
    constexpr int WHALVES = KT * NT * 32 * 16;     // halves per packed matrix

    __shared__ _Float16 sW[2 * WHALVES];           // [Wgc | Wbi], <=16 KB

    int lane = threadIdx.x & 31;
    int nl   = lane & 15;                 // A row / B,C column within tile
    int kh   = lane >> 4;                 // K-half selector
    int wave = (blockIdx.x * blockDim.x + threadIdx.x) >> 5;
    int m0   = wave * 16;

    // ---- hoisted bias loads (overlap the LDS fill) ----
    float bg[NT], bb[NT];
#pragma unroll
    for (int t = 0; t < NT; ++t) {
        bg[t] = bgc[t * 16 + nl];
        bb[t] = bbi[t * 16 + nl];
    }

    // ---- cooperative LDS staging of packed weights (uint4 = 8 halves) ----
    {
        constexpr int NV4 = WHALVES / 8;
        const uint4* sg = (const uint4*)Wgc;
        const uint4* sb = (const uint4*)Wbi;
        uint4* dg = (uint4*)sW;
        uint4* db = (uint4*)(sW + WHALVES);
        for (int i = threadIdx.x; i < NV4; i += 256) dg[i] = sg[i];
        for (int i = threadIdx.x; i < NV4; i += 256) db[i] = sb[i];
    }
    __syncthreads();

    if (m0 >= N_NODES) return;            // uniform per wave -> EXEC all-1 below

    int arow = m0 + nl;                   // A-matrix row this lane loads

    // ---- build A fragments: sum = x+side, bi = x*side, converted to f16 ----
    v16h a_sum[KT], a_bi[KT];
    const float* xr = x + (size_t)arow * DIN;
    const float* sr = side + (size_t)arow * DIN;
#pragma unroll
    for (int kt = 0; kt < KT; ++kt) {
        int k0 = kt * 32 + kh * 8;
        float xv[16], sv[16];
        *(float4*)&xv[0]  = *(const float4*)(xr + k0);
        *(float4*)&xv[4]  = *(const float4*)(xr + k0 + 4);
        *(float4*)&xv[8]  = *(const float4*)(xr + k0 + 16);
        *(float4*)&xv[12] = *(const float4*)(xr + k0 + 20);
        *(float4*)&sv[0]  = *(const float4*)(sr + k0);
        *(float4*)&sv[4]  = *(const float4*)(sr + k0 + 4);
        *(float4*)&sv[8]  = *(const float4*)(sr + k0 + 16);
        *(float4*)&sv[12] = *(const float4*)(sr + k0 + 20);
        v16h as, ab;
#pragma unroll
        for (int e = 0; e < 16; ++e) {
            as[e] = (_Float16)(xv[e] + sv[e]);
            ab[e] = (_Float16)(xv[e] * sv[e]);
        }
        a_sum[kt] = as;
        a_bi[kt]  = ab;
    }

    // ---- WMMA per n-tile (B from LDS), then fused leaky-relu + add ----
    v8f outv[NT];
#pragma unroll
    for (int t = 0; t < NT; ++t) {
        v8f cg, cb;
#pragma unroll
        for (int r = 0; r < 8; ++r) { cg[r] = bg[t]; cb[r] = bb[t]; }
#pragma unroll
        for (int kt = 0; kt < KT; ++kt) {
            v16h Bg = *(const v16h*)(sW + ((size_t)(kt * NT + t) * 32 + lane) * 16);
            v16h Bb = *(const v16h*)(sW + WHALVES +
                                     ((size_t)(kt * NT + t) * 32 + lane) * 16);
            cg = __builtin_amdgcn_wmma_f32_16x16x32_f16(
                    false, a_sum[kt], false, Bg, (short)0, cg, false, false);
            cb = __builtin_amdgcn_wmma_f32_16x16x32_f16(
                    false, a_bi[kt], false, Bb, (short)0, cb, false, false);
        }
        v8f o;
#pragma unroll
        for (int r = 0; r < 8; ++r) {
            float g = cg[r]; g = fmaxf(g, 0.01f * g);   // leaky_relu, slope 0.01
            float b = cb[r]; b = fmaxf(b, 0.01f * b);
            o[r] = g + b;
        }
        outv[t] = o;
    }

    // ---- row L2 norm: reduce over n (16 lanes within each half + n-tiles) ----
    float scl[8];
#pragma unroll
    for (int r = 0; r < 8; ++r) {
        float ss = 0.0f;
#pragma unroll
        for (int t = 0; t < NT; ++t) ss += outv[t][r] * outv[t][r];
        ss += __shfl_xor(ss, 1, 32);
        ss += __shfl_xor(ss, 2, 32);
        ss += __shfl_xor(ss, 4, 32);
        ss += __shfl_xor(ss, 8, 32);
        scl[r] = 1.0f / fmaxf(sqrtf(ss), 1e-12f);
    }

    // ---- write x_next (dense DIN_next=DOUT) and out slice (stride 176) ----
#pragma unroll
    for (int t = 0; t < NT; ++t) {
#pragma unroll
        for (int r = 0; r < 8; ++r) {
            int gm = m0 + r + 8 * kh;
            int gn = t * 16 + nl;
            float v = outv[t][r] * scl[r];
            x_next[(size_t)gm * DOUT + gn]   = v;
            out[(size_t)gm * OUT_COLS + gn]  = v;
        }
    }
}

// ---------------------------------------------------------------------------
extern "C" void kernel_launch(void* const* d_in, const int* in_sizes, int n_in,
                              void* d_out, int out_size, void* d_ws, size_t ws_size,
                              hipStream_t stream)
{
    const float* ue   = (const float*)d_in[0];
    const float* ee   = (const float*)d_in[1];
    const float* ev   = (const float*)d_in[2];
    const int*   erow = (const int*)d_in[3];
    const int*   ecol = (const int*)d_in[4];
    const float* Wgc0 = (const float*)d_in[5];
    const float* bgc0 = (const float*)d_in[6];
    const float* Wbi0 = (const float*)d_in[7];
    const float* bbi0 = (const float*)d_in[8];
    const float* Wgc1 = (const float*)d_in[9];
    const float* bgc1 = (const float*)d_in[10];
    const float* Wbi1 = (const float*)d_in[11];
    const float* bbi1 = (const float*)d_in[12];
    const float* Wgc2 = (const float*)d_in[13];
    const float* bgc2 = (const float*)d_in[14];
    const float* Wbi2 = (const float*)d_in[15];
    const float* bbi2 = (const float*)d_in[16];
    float* out = (float*)d_out;

    // workspace carve-up
    float* x_a  = (float*)d_ws;                     // N_NODES*64 f32
    float* x_b  = x_a + (size_t)N_NODES * 64;       // N_NODES*64 f32
    float* side = x_b + (size_t)N_NODES * 64;       // N_NODES*64 f32
    _Float16* wp = (_Float16*)(side + (size_t)N_NODES * 64);
    const size_t FRAG = 2 * 4 * 32 * 16;            // max halves per matrix (8 KB)
    _Float16* pgc0 = wp + 0 * FRAG;
    _Float16* pbi0 = wp + 1 * FRAG;
    _Float16* pgc1 = wp + 2 * FRAG;
    _Float16* pbi1 = wp + 3 * FRAG;
    _Float16* pgc2 = wp + 4 * FRAG;
    _Float16* pbi2 = wp + 5 * FRAG;

    // init x0 + output cols [0,64)
    init_x_kernel<<<(N_NODES * 16 + 255) / 256, 256, 0, stream>>>(ue, ee, x_a, out);

    // pack weights into WMMA B-fragment layout
    pack_weights_kernel<<<2 * 4, 32, 0, stream>>>(Wgc0, pgc0, 64, 64);
    pack_weights_kernel<<<2 * 4, 32, 0, stream>>>(Wbi0, pbi0, 64, 64);
    pack_weights_kernel<<<2 * 2, 32, 0, stream>>>(Wgc1, pgc1, 64, 32);
    pack_weights_kernel<<<2 * 2, 32, 0, stream>>>(Wbi1, pbi1, 64, 32);
    pack_weights_kernel<<<1 * 1, 32, 0, stream>>>(Wgc2, pgc2, 32, 16);
    pack_weights_kernel<<<1 * 1, 32, 0, stream>>>(Wbi2, pbi2, 32, 16);

    const int waves      = N_NODES / 16;                 // 9375, N divisible by 16
    const int gemm_grid  = (waves + 7) / 8;              // 8 waves / 256-thread block

    // ---- layer 0: din=64 dout=64, out cols [64,128) ----
    hipMemsetAsync(side, 0, (size_t)N_NODES * 64 * sizeof(float), stream);
    spmm_scatter_kernel<64><<<(N_EDGES * 16 + 255) / 256, 256, 0, stream>>>(
        x_a, ev, erow, ecol, side);
    layer_kernel<64, 64><<<gemm_grid, 256, 0, stream>>>(
        x_a, side, pgc0, bgc0, pbi0, bbi0, x_b, out + 64);

    // ---- layer 1: din=64 dout=32, out cols [128,160) ----
    hipMemsetAsync(side, 0, (size_t)N_NODES * 64 * sizeof(float), stream);
    spmm_scatter_kernel<64><<<(N_EDGES * 16 + 255) / 256, 256, 0, stream>>>(
        x_b, ev, erow, ecol, side);
    layer_kernel<64, 32><<<gemm_grid, 256, 0, stream>>>(
        x_b, side, pgc1, bgc1, pbi1, bbi1, x_a, out + 128);

    // ---- layer 2: din=32 dout=16, out cols [160,176) ----
    hipMemsetAsync(side, 0, (size_t)N_NODES * 32 * sizeof(float), stream);
    spmm_scatter_kernel<32><<<(N_EDGES * 8 + 255) / 256, 256, 0, stream>>>(
        x_a, ev, erow, ecol, side);
    layer_kernel<32, 16><<<gemm_grid, 256, 0, stream>>>(
        x_a, side, pgc2, bgc2, pbi2, bbi2, x_b, out + 160);
}
